// LSTM_Encoder_14645838479698
// MI455X (gfx1250) — matrix operand order repaired and checked by reading, not compiled, not run on hardware
//
#include <hip/hip_runtime.h>

// Problem constants (match reference).
#define EMBED    512
#define HIDDEN   512
#define BATCH    64
#define SEQ      512
#define GATES    2048          // 4*HIDDEN, torch gate order i,f,g,o
#define MTILE    16            // batch rows per workgroup (one WMMA M tile)
#define NTHREADS 512           // 16 wave32
#define LSTRIDE  520           // padded row stride (bf16 elems) for LDS tiles
#define HBUF     (MTILE * LSTRIDE)

typedef __attribute__((ext_vector_type(16))) __bf16 v16bf;
typedef __attribute__((ext_vector_type(8)))  float  v8f;

union AFrag { v16bf v; uint4 q[2]; };

__device__ __forceinline__ unsigned short f32_to_bf16_rne(float f) {
  unsigned u = __float_as_uint(f);
  unsigned r = u + 0x7FFFu + ((u >> 16) & 1u);
  return (unsigned short)(r >> 16);
}

// Raw v_rcp_f32 (1ulp) instead of the IEEE division expansion: the gate math
// sits on the serial recurrence's critical path and feeds a bf16 GEMM, so the
// ~10-instruction div_scale/fma/fixup chain is pure latency waste.
__device__ __forceinline__ float fast_rcp(float x) { return __builtin_amdgcn_rcpf(x); }
__device__ __forceinline__ float sigm(float x) { return fast_rcp(1.0f + __expf(-x)); }
__device__ __forceinline__ float tanh_f(float x) {
  return __builtin_fmaf(2.0f, fast_rcp(1.0f + __expf(-2.0f * x)), -1.0f);
}

// ---------------------------------------------------------------------------
// Pre-block weights: f32 [2048][512] row-major -> bf16 WMMA-B fragment blocks.
// Block (n_t,k_t): N=[n_t*16,+16), K=[k_t*32,+32). Lane l<16 -> N=n_t*16+l,
// K=k_t*32+{0..7,16..23}; l>=16 -> N=n_t*16+l-16, K=k_t*32+{8..15,24..31}; 16 bf16
// (32B) per lane. Block = 1KB -> one fully coalesced pair of b128s per lane.
// ---------------------------------------------------------------------------
__global__ void convert_weights(const float* __restrict__ W,
                                unsigned short* __restrict__ out, int total) {
  for (int e = blockIdx.x * blockDim.x + threadIdx.x; e < total;
       e += gridDim.x * blockDim.x) {
    int j   = e & 15;
    int l   = (e >> 4) & 31;
    int blk = e >> 9;
    int k_t = blk & 15;        // EMBED/32 = 16 k-tiles
    int n_t = blk >> 4;
    int n = n_t * 16 + (l & 15);
    int k = k_t * 32 + (j >> 3) * 16 + (l >> 4) * 8 + (j & 7);
    out[e] = f32_to_bf16_rne(W[n * EMBED + k]);
  }
}

// ---------------------------------------------------------------------------
// Persistent per-batch-slice LSTM. One workgroup owns 16 batch rows for the
// whole sequence; wave w owns hidden chunk [w*32, w*32+32), processed as two
// 16-wide sub-chunks (s=0,1). For each sub-chunk the wave computes the four
// gate tiles (i,f,g,o) with a full k-sweep (only 4 accumulators live), then
// immediately applies the gate math: c stays in registers, new h goes to the
// ping-pong LDS buffer for the next step's A operand. No inter-workgroup
// synchronization exists anywhere in the 512-step recurrence.
// ---------------------------------------------------------------------------
__global__ void __launch_bounds__(NTHREADS, 1)
lstm_persistent(const int* __restrict__ tokens, const float* __restrict__ emb,
                const float* __restrict__ b_ih, const float* __restrict__ b_hh,
                const unsigned short* __restrict__ Wih,   // blocked bf16
                const unsigned short* __restrict__ Whh,   // blocked bf16
                float* __restrict__ out) {
  __shared__ __align__(16) unsigned short x_lds[HBUF];
  __shared__ __align__(16) unsigned short h_lds[2 * HBUF];  // ping-pong

  const int tid  = threadIdx.x;
  const int lane = tid & 31;
  const int wv   = tid >> 5;            // wave 0..15 (wave32)
  const int b0   = blockIdx.x * MTILE;  // batch base

  // h0 = 0 (zero both buffers; only buffer 0 is read at t=0)
  for (int i = tid; i < 2 * HBUF; i += NTHREADS) h_lds[i] = 0;

  // Per-lane bias (b_ih + b_hh) for each of this wave's 8 N-tiles.
  float bias[4][2];
#pragma unroll
  for (int g = 0; g < 4; ++g)
#pragma unroll
    for (int s = 0; s < 2; ++s) {
      int n = g * HIDDEN + wv * 32 + s * 16 + (lane & 15);
      bias[g][s] = b_ih[n] + b_hh[n];
    }

  v8f c_st[2];
#pragma unroll
  for (int s = 0; s < 2; ++s)
#pragma unroll
    for (int r = 0; r < 8; ++r) c_st[s][r] = 0.0f;

  // x loader: thread -> (row = tid>>5, 16 columns at (tid&31)*16)
  const int xrow = tid >> 5;
  const int xcol = (tid & 31) * 16;
  auto load_x = [&](int t) {
    int tok = tokens[(b0 + xrow) * SEQ + t];
    const float4* src = (const float4*)(emb + (size_t)tok * EMBED + xcol);
    union { uint4 q; unsigned short us[8]; } pk[2];
#pragma unroll
    for (int q4 = 0; q4 < 4; ++q4) {
      float4 v = src[q4];
      pk[q4 >> 1].us[(q4 & 1) * 4 + 0] = f32_to_bf16_rne(v.x);
      pk[q4 >> 1].us[(q4 & 1) * 4 + 1] = f32_to_bf16_rne(v.y);
      pk[q4 >> 1].us[(q4 & 1) * 4 + 2] = f32_to_bf16_rne(v.z);
      pk[q4 >> 1].us[(q4 & 1) * 4 + 3] = f32_to_bf16_rne(v.w);
    }
    uint4* dst = (uint4*)&x_lds[xrow * LSTRIDE + xcol];
    dst[0] = pk[0].q;
    dst[1] = pk[1].q;
  };

  load_x(0);
  __syncthreads();

  // A-fragment addressing (16-bit A layout): lane<16 -> M=lane, K=c..c+7 and
  // c+16..c+23 with c=0; lane>=16 -> M=lane-16, c=8.
  const int a_base = (lane & 15) * LSTRIDE + (lane >> 4) * 8;
  // Weight byte offset: block (n>>4 = g*32 + wv*2 + s, kt) -> bytes =
  // g*524288 + wv*32768 + s*16384 + kt*1024 + lane*32.
  const size_t wbase = (size_t)wv * 32768 + (size_t)lane * 32;
  const char* wx_base = (const char*)Wih + wbase;
  const char* wh_base = (const char*)Whh + wbase;

  for (int t = 0; t < SEQ; ++t) {
    const int cur = (t & 1) * HBUF;
    const int nxt = HBUF - cur;

#pragma unroll
    for (int s = 0; s < 2; ++s) {
      v8f a0, a1, a2, a3;  // gate accumulators i, f, g, o for this sub-chunk
#pragma unroll
      for (int r = 0; r < 8; ++r) {
        a0[r] = bias[0][s]; a1[r] = bias[1][s];
        a2[r] = bias[2][s]; a3[r] = bias[3][s];
      }

#pragma unroll 1
      for (int k0 = 0; k0 < EMBED; k0 += 32) {
        AFrag ax, ah;
        ax.q[0] = *(const uint4*)&x_lds[a_base + k0];
        ax.q[1] = *(const uint4*)&x_lds[a_base + k0 + 16];
        ah.q[0] = *(const uint4*)&h_lds[cur + a_base + k0];
        ah.q[1] = *(const uint4*)&h_lds[cur + a_base + k0 + 16];

        const char* wx = wx_base + (size_t)s * 16384 + (size_t)(k0 >> 5) * 1024;
        const char* wh = wh_base + (size_t)s * 16384 + (size_t)(k0 >> 5) * 1024;

        AFrag b;
        // gate 0 (i)
        b.q[0] = *(const uint4*)(wx + 0 * 524288);
        b.q[1] = *(const uint4*)(wx + 0 * 524288 + 16);
        a0 = __builtin_amdgcn_wmma_f32_16x16x32_bf16(false, ax.v, false, b.v, (short)0, a0, false, false);
        b.q[0] = *(const uint4*)(wh + 0 * 524288);
        b.q[1] = *(const uint4*)(wh + 0 * 524288 + 16);
        a0 = __builtin_amdgcn_wmma_f32_16x16x32_bf16(false, ah.v, false, b.v, (short)0, a0, false, false);
        // gate 1 (f)
        b.q[0] = *(const uint4*)(wx + 1 * 524288);
        b.q[1] = *(const uint4*)(wx + 1 * 524288 + 16);
        a1 = __builtin_amdgcn_wmma_f32_16x16x32_bf16(false, ax.v, false, b.v, (short)0, a1, false, false);
        b.q[0] = *(const uint4*)(wh + 1 * 524288);
        b.q[1] = *(const uint4*)(wh + 1 * 524288 + 16);
        a1 = __builtin_amdgcn_wmma_f32_16x16x32_bf16(false, ah.v, false, b.v, (short)0, a1, false, false);
        // gate 2 (g)
        b.q[0] = *(const uint4*)(wx + 2 * 524288);
        b.q[1] = *(const uint4*)(wx + 2 * 524288 + 16);
        a2 = __builtin_amdgcn_wmma_f32_16x16x32_bf16(false, ax.v, false, b.v, (short)0, a2, false, false);
        b.q[0] = *(const uint4*)(wh + 2 * 524288);
        b.q[1] = *(const uint4*)(wh + 2 * 524288 + 16);
        a2 = __builtin_amdgcn_wmma_f32_16x16x32_bf16(false, ah.v, false, b.v, (short)0, a2, false, false);
        // gate 3 (o)
        b.q[0] = *(const uint4*)(wx + 3 * 524288);
        b.q[1] = *(const uint4*)(wx + 3 * 524288 + 16);
        a3 = __builtin_amdgcn_wmma_f32_16x16x32_bf16(false, ax.v, false, b.v, (short)0, a3, false, false);
        b.q[0] = *(const uint4*)(wh + 3 * 524288);
        b.q[1] = *(const uint4*)(wh + 3 * 524288 + 16);
        a3 = __builtin_amdgcn_wmma_f32_16x16x32_bf16(false, ah.v, false, b.v, (short)0, a3, false, false);
      }

      // Gate math for this 16-wide hidden chunk, in WMMA C layout:
      // element (r, lane) -> M = r + 8*(lane>=16), N = wv*32+s*16+(lane&15).
#pragma unroll
      for (int r = 0; r < 8; ++r) {
        float iv = sigm(a0[r]);
        float fv = sigm(a1[r]);
        float gv = tanh_f(a2[r]);
        float ov = sigm(a3[r]);
        float cv = __builtin_fmaf(fv, c_st[s][r], iv * gv);
        c_st[s][r] = cv;
        float hv = ov * tanh_f(cv);
        int m = r + ((lane >> 4) << 3);
        int j = wv * 32 + s * 16 + (lane & 15);
        h_lds[nxt + m * LSTRIDE + j] = f32_to_bf16_rne(hv);
        if (t == SEQ - 1) out[(size_t)(b0 + m) * HIDDEN + j] = hv;
      }
    }

    __syncthreads();               // all waves done reading x_lds / h_lds[cur]
    if (t + 1 < SEQ) load_x(t + 1);
    __syncthreads();               // x_{t+1} and h_lds[nxt] ready
  }

  // Emit c ([64,512] f32 after h).
#pragma unroll
  for (int s = 0; s < 2; ++s)
#pragma unroll
    for (int r = 0; r < 8; ++r) {
      int m = r + ((lane >> 4) << 3);
      int j = wv * 32 + s * 16 + (lane & 15);
      out[(size_t)BATCH * HIDDEN + (size_t)(b0 + m) * HIDDEN + j] = c_st[s][r];
    }
}

// ---------------------------------------------------------------------------
extern "C" void kernel_launch(void* const* d_in, const int* in_sizes, int n_in,
                              void* d_out, int out_size, void* d_ws, size_t ws_size,
                              hipStream_t stream) {
  const int*   tokens = (const int*)d_in[0];
  const float* emb    = (const float*)d_in[1];
  const float* W_ih   = (const float*)d_in[2];
  const float* W_hh   = (const float*)d_in[3];
  const float* b_ih   = (const float*)d_in[4];
  const float* b_hh   = (const float*)d_in[5];
  float*       out    = (float*)d_out;

  unsigned short* wih_b = (unsigned short*)d_ws;                 // 2 MB
  unsigned short* whh_b = wih_b + (size_t)GATES * EMBED;         // 2 MB
  const int total = GATES * EMBED;

  convert_weights<<<1024, 256, 0, stream>>>(W_ih, wih_b, total);
  convert_weights<<<1024, 256, 0, stream>>>(W_hh, whh_b, total);
  lstm_persistent<<<BATCH / MTILE, NTHREADS, 0, stream>>>(
      tokens, emb, b_ih, b_hh, wih_b, whh_b, out);
}
